// MambaLMHeadModel_498216206634
// MI455X (gfx1250) — compile-verified, matrix operand order checked by reading, not run on hardware
//
#include <hip/hip_runtime.h>
#include <hip/hip_bf16.h>
#include <math.h>

#define D_MODEL 768
#define D_INNER 1536
#define D_STATE 16
#define D_CONV  4
#define DT_RANK 48
#define D_MLP   2048
#define VOCAB   50304
#define NTOK    1024        // B*T = 2*512
#define SEQ     512
#define NLAYER  4
#define DBC_LD  128         // padded row stride of dbc

typedef __attribute__((ext_vector_type(16))) __bf16 v16bf;
typedef __attribute__((ext_vector_type(8)))  float  v8f;

union FragBF { v16bf v; unsigned int u[8]; };

__device__ __forceinline__ unsigned short f2bf(float f) {
    unsigned int x = __float_as_uint(f);
    unsigned int r = x + 0x7FFFu + ((x >> 16) & 1u);   // round-to-nearest-even
    return (unsigned short)(r >> 16);
}

__device__ __forceinline__ float siluf(float v) {
    return v / (1.0f + __expf(-v));
}

// ---------------------------------------------------------------------------
// Fast WMMA GEMM: C[1024,N] = A[1024,K] * B^T, all dims exactly tiled.
//   Requires: N % 128 == 0, K % 64 == 0, lda % 4 == 0 (16B-aligned rows).
//   B source is [N,K] (k-contiguous): bf16 pre-transposed weights (BF32=false,
//   staged with global_load_async_to_lds_b128 / ASYNCcnt) or raw f32
//   (BF32=true, converted in-flight; used for the LM head).
// Block: 256 threads = 8 waves; tile 128x128; K-step 64.
// Waves: 4(M) x 2(N); each wave 32x64 = 2x4 WMMA frags -> 16 WMMA / K-iter.
// ---------------------------------------------------------------------------
#define BM 128
#define BN 128
#define BK 64
#define LSTR2 72   // halfword row stride in LDS (144B rows: every row 16B-aligned)

template<bool BF32>
__global__ __launch_bounds__(256) void gemm_wmma_t(
        const float* __restrict__ A, const void* __restrict__ Bsrc,
        float* __restrict__ C, int N, int K, int lda, int ldc)
{
    __shared__ unsigned short sA[BM * LSTR2];   // [m][k]
    __shared__ unsigned short sB[BN * LSTR2];   // [n][k]

    const int tid  = threadIdx.x;
    const int lane = tid & 31;
    const int wave = tid >> 5;
    const int wm   = (wave >> 1) * 32;      // 0,32,64,96
    const int wn   = (wave & 1) * 64;       // 0,64
    const int tileM = blockIdx.y * BM;
    const int tileN = blockIdx.x * BN;

    v8f acc[2][4] = {};

    const int mrow = lane & 15;
    const int kb   = (lane >> 4) << 3;      // 0 or 8 (ISA 16-bit A/B layout)

    for (int kk = 0; kk < K; kk += BK) {
        // ---- stage A: 128 rows x 64 k, f32 -> bf16, float4 loads ----
        #pragma unroll
        for (int i = tid; i < BM * BK / 4; i += 256) {
            int r = i >> 4, c4 = (i & 15) << 2;
            float4 v = *(const float4*)(A + (size_t)(tileM + r) * lda + kk + c4);
            uint2 st;
            st.x = (unsigned int)f2bf(v.x) | ((unsigned int)f2bf(v.y) << 16);
            st.y = (unsigned int)f2bf(v.z) | ((unsigned int)f2bf(v.w) << 16);
            *(uint2*)(&sA[r * LSTR2 + c4]) = st;
        }
        // ---- stage B: 128 n-rows x 64 k ----
        if (BF32) {
            const float* B = (const float*)Bsrc;
            #pragma unroll
            for (int i = tid; i < BN * BK / 4; i += 256) {
                int r = i >> 4, c4 = (i & 15) << 2;
                float4 v = *(const float4*)(B + (size_t)(tileN + r) * K + kk + c4);
                uint2 st;
                st.x = (unsigned int)f2bf(v.x) | ((unsigned int)f2bf(v.y) << 16);
                st.y = (unsigned int)f2bf(v.z) | ((unsigned int)f2bf(v.w) << 16);
                *(uint2*)(&sB[r * LSTR2 + c4]) = st;
            }
        } else {
            // async copy: 16B per lane straight into LDS, tracked by ASYNCcnt
            const unsigned short* B = (const unsigned short*)Bsrc;
            #pragma unroll
            for (int i = tid; i < BN * BK / 8; i += 256) {
                int r = i >> 3, c8 = (i & 7) << 3;
                const unsigned short* gp = B + (size_t)(tileN + r) * K + kk + c8;
                unsigned int ldsoff =
                    (unsigned int)(unsigned long long)(const void*)&sB[r * LSTR2 + c8];
                asm volatile("global_load_async_to_lds_b128 %0, %1, off"
                             :: "v"(ldsoff), "v"(gp) : "memory");
            }
        }
        // prefetch next K-slab (emits global_prefetch_b8)
        if (kk + BK < K) {
            __builtin_prefetch(A + (size_t)(tileM + (tid >> 1)) * lda + kk + BK, 0, 1);
            if (BF32)
                __builtin_prefetch((const float*)Bsrc + (size_t)(tileN + (tid >> 1)) * K + kk + BK, 0, 1);
            else
                __builtin_prefetch((const unsigned short*)Bsrc + (size_t)(tileN + (tid >> 1)) * K + kk + BK, 0, 1);
        }
        if (!BF32)
            asm volatile("s_wait_asynccnt 0x0" ::: "memory");
        __syncthreads();

        // ---- two 16x16x32 sub-steps over the 64-wide K slab ----
        #pragma unroll
        for (int ks = 0; ks < 2; ++ks) {
            FragBF fa[2], fb[4];
            #pragma unroll
            for (int i = 0; i < 2; ++i) {
                const unsigned short* ra = &sA[(wm + i * 16 + mrow) * LSTR2 + ks * 32];
                #pragma unroll
                for (int p = 0; p < 4; ++p) {
                    fa[i].u[p]     = *(const unsigned int*)(ra + kb + 2 * p);
                    fa[i].u[4 + p] = *(const unsigned int*)(ra + 16 + kb + 2 * p);
                }
            }
            #pragma unroll
            for (int j = 0; j < 4; ++j) {
                const unsigned short* rb = &sB[(wn + j * 16 + mrow) * LSTR2 + ks * 32];
                #pragma unroll
                for (int p = 0; p < 4; ++p) {
                    fb[j].u[p]     = *(const unsigned int*)(rb + kb + 2 * p);
                    fb[j].u[4 + p] = *(const unsigned int*)(rb + 16 + kb + 2 * p);
                }
            }
            #pragma unroll
            for (int i = 0; i < 2; ++i)
                #pragma unroll
                for (int j = 0; j < 4; ++j)
                    acc[i][j] = __builtin_amdgcn_wmma_f32_16x16x32_bf16(
                            false, fa[i].v, false, fb[j].v,
                            (short)0, acc[i][j], false, false);
        }
        __syncthreads();
    }

    // C/D layout: lane&15 = N, (lane>=16 ? 8:0)+vgpr = M
    const int nofs = lane & 15;
    const int mofs = (lane >> 4) * 8;
    #pragma unroll
    for (int i = 0; i < 2; ++i)
        #pragma unroll
        for (int j = 0; j < 4; ++j) {
            int gn = tileN + wn + j * 16 + nofs;
            #pragma unroll
            for (int r = 0; r < 8; ++r) {
                int gm = tileM + wm + i * 16 + mofs + r;
                C[(size_t)gm * ldc + gn] = acc[i][j][r];
            }
        }
}

// ---------------------------------------------------------------------------
// Weight transpose+convert: Bt[n*Kd + k] = bf16(B[k*Ns + n]) with zero pad.
// ---------------------------------------------------------------------------
__global__ void convert_wT_k(const float* __restrict__ B, unsigned short* __restrict__ Bt,
                             int Ks, int Ns, int Kd, int Nd)
{
    int i = blockIdx.x * 256 + threadIdx.x;
    if (i >= Kd * Nd) return;
    int n = i / Kd;
    int k = i - n * Kd;
    float v = (k < Ks && n < Ns) ? B[(size_t)k * Ns + n] : 0.0f;
    Bt[i] = f2bf(v);
}

// ---------------------------------------------------------------------------
// Embedding gather
// ---------------------------------------------------------------------------
__global__ void embed_k(const int* __restrict__ idx, const float* __restrict__ emb,
                        float* __restrict__ hidden)
{
    int row = blockIdx.x;
    int tok = idx[row];
    if (tok < 0 || tok >= VOCAB) tok = 0;
    for (int c = threadIdx.x; c < D_MODEL; c += blockDim.x)
        hidden[(size_t)row * D_MODEL + c] = emb[(size_t)tok * D_MODEL + c];
}

// ---------------------------------------------------------------------------
// residual = first ? hidden : hidden + residual ; xn = rmsnorm(residual) * w
// ---------------------------------------------------------------------------
__global__ __launch_bounds__(256) void add_rmsnorm_k(
        const float* __restrict__ hidden, float* __restrict__ residual,
        const float* __restrict__ w, float* __restrict__ xn, int first)
{
    __shared__ float red[8];
    int row = blockIdx.x;
    int tid = threadIdx.x;
    float vals[3];
    float local = 0.0f;
    #pragma unroll
    for (int q = 0; q < 3; ++q) {
        int c = tid + q * 256;
        float v = hidden[(size_t)row * D_MODEL + c];
        if (!first) v += residual[(size_t)row * D_MODEL + c];
        residual[(size_t)row * D_MODEL + c] = v;
        vals[q] = v;
        local += v * v;
    }
    for (int o = 16; o > 0; o >>= 1) local += __shfl_down(local, o, 32);
    if ((tid & 31) == 0) red[tid >> 5] = local;
    __syncthreads();
    float tot = 0.0f;
    #pragma unroll
    for (int q = 0; q < 8; ++q) tot += red[q];
    float scale = rsqrtf(tot / (float)D_MODEL + 1e-5f);
    #pragma unroll
    for (int q = 0; q < 3; ++q) {
        int c = tid + q * 256;
        xn[(size_t)row * D_MODEL + c] = vals[q] * scale * w[c];
    }
}

// ---------------------------------------------------------------------------
// Causal depthwise conv (width 4) + bias + SiLU
// ---------------------------------------------------------------------------
__global__ void conv_silu_k(const float* __restrict__ xz, const float* __restrict__ cw,
                            const float* __restrict__ cb, float* __restrict__ xs)
{
    int i = blockIdx.x * blockDim.x + threadIdx.x;
    if (i >= NTOK * D_INNER) return;
    int c   = i % D_INNER;
    int row = i / D_INNER;
    int t   = row & (SEQ - 1);
    float acc = cb[c];
    #pragma unroll
    for (int k = 0; k < D_CONV; ++k) {
        int tt = t + k - (D_CONV - 1);
        if (tt >= 0)
            acc += xz[(size_t)(row + k - (D_CONV - 1)) * (2 * D_INNER) + c] * cw[c * D_CONV + k];
    }
    xs[i] = siluf(acc);
}

// ---------------------------------------------------------------------------
// Selective scan
// ---------------------------------------------------------------------------
__global__ __launch_bounds__(256) void scan_k(
        const float* __restrict__ dtpre, const float* __restrict__ dt_b,
        const float* __restrict__ A_log, const float* __restrict__ xs,
        const float* __restrict__ xz,    const float* __restrict__ dbc,
        const float* __restrict__ Dv,    float* __restrict__ y)
{
    __shared__ float sB[D_STATE], sC[D_STATE];
    int e = blockIdx.x * 256 + threadIdx.x;
    int b = blockIdx.y;
    float Areg[D_STATE], h[D_STATE];
    #pragma unroll
    for (int n = 0; n < D_STATE; ++n) {
        Areg[n] = -__expf(A_log[(size_t)e * D_STATE + n]);
        h[n] = 0.0f;
    }
    float dtb = dt_b[e];
    float dco = Dv[e];
    for (int t = 0; t < SEQ; ++t) {
        int row = b * SEQ + t;
        if (threadIdx.x < 2 * D_STATE) {
            float v = dbc[(size_t)row * DBC_LD + DT_RANK + threadIdx.x];
            if (threadIdx.x < D_STATE) sB[threadIdx.x] = v;
            else                       sC[threadIdx.x - D_STATE] = v;
        }
        __syncthreads();
        float dt = dtpre[(size_t)row * D_INNER + e] + dtb;
        dt = (dt > 20.0f) ? dt : log1pf(__expf(dt));
        float xv = xs[(size_t)row * D_INNER + e];
        float acc = 0.0f;
        #pragma unroll
        for (int n = 0; n < D_STATE; ++n) {
            float dA = __expf(dt * Areg[n]);
            h[n] = dA * h[n] + dt * sB[n] * xv;
            acc += h[n] * sC[n];
        }
        acc += dco * xv;
        float z = xz[(size_t)row * (2 * D_INNER) + D_INNER + e];
        y[(size_t)row * D_INNER + e] = acc * siluf(z);
        __syncthreads();
    }
}

// ---------------------------------------------------------------------------
// GLU: act = u * silu(gate)
// ---------------------------------------------------------------------------
__global__ void glu_k(const float* __restrict__ fc, float* __restrict__ act)
{
    int i = blockIdx.x * blockDim.x + threadIdx.x;
    if (i >= NTOK * D_MLP) return;
    int row = i >> 11;
    int c   = i & (D_MLP - 1);
    float u = fc[(size_t)row * (2 * D_MLP) + c];
    float g = fc[(size_t)row * (2 * D_MLP) + D_MLP + c];
    act[i] = u * siluf(g);
}

// ---------------------------------------------------------------------------
// Cross-entropy
// ---------------------------------------------------------------------------
__global__ void loss_zero_k(float* loss) { if (threadIdx.x == 0) *loss = 0.0f; }

__global__ __launch_bounds__(256) void loss_k(
        const float* __restrict__ logits, const int* __restrict__ targets,
        float* __restrict__ loss)
{
    __shared__ float red[8];
    int row = blockIdx.x;
    const float* lr = logits + (size_t)row * VOCAB;
    float mx = -3.0e38f;
    for (int c = threadIdx.x; c < VOCAB; c += 256) mx = fmaxf(mx, lr[c]);
    for (int o = 16; o > 0; o >>= 1) mx = fmaxf(mx, __shfl_down(mx, o, 32));
    if ((threadIdx.x & 31) == 0) red[threadIdx.x >> 5] = mx;
    __syncthreads();
    float m = red[0];
    #pragma unroll
    for (int q = 1; q < 8; ++q) m = fmaxf(m, red[q]);
    __syncthreads();
    float s = 0.0f;
    for (int c = threadIdx.x; c < VOCAB; c += 256) s += __expf(lr[c] - m);
    for (int o = 16; o > 0; o >>= 1) s += __shfl_down(s, o, 32);
    if ((threadIdx.x & 31) == 0) red[threadIdx.x >> 5] = s;
    __syncthreads();
    if (threadIdx.x == 0) {
        float tot = 0.0f;
        #pragma unroll
        for (int q = 0; q < 8; ++q) tot += red[q];
        int tv = targets[row];
        if (tv != -1) {
            int tc = tv < 0 ? 0 : (tv >= VOCAB ? VOCAB - 1 : tv);
            float nll = (m + __logf(tot)) - lr[tc];
            atomicAdd(loss, nll * (1.0f / (float)NTOK));
        }
    }
}

// ---------------------------------------------------------------------------
// Host launcher
// ---------------------------------------------------------------------------
static inline void cvtT(const float* W, unsigned short* Bt, int Ks, int Ns,
                        int Kd, int Nd, hipStream_t stream)
{
    int n = Kd * Nd;
    convert_wT_k<<<(n + 255) / 256, 256, 0, stream>>>(W, Bt, Ks, Ns, Kd, Nd);
}

extern "C" void kernel_launch(void* const* d_in, const int* in_sizes, int n_in,
                              void* d_out, int out_size, void* d_ws, size_t ws_size,
                              hipStream_t stream)
{
    (void)in_sizes; (void)n_in; (void)ws_size;

    const int*   idx     = (const int*)d_in[0];
    const int*   targets = (const int*)d_in[1];
    const float* emb     = (const float*)d_in[2];
    const float* norm_f  = (const float*)d_in[3];

    // workspace layout
    float* ws = (float*)d_ws;
    float* hidden   = ws;                     ws += NTOK * D_MODEL;
    float* residual = ws;                     ws += NTOK * D_MODEL;
    float* xn       = ws;                     ws += NTOK * D_MODEL;
    float* xz       = ws;                     ws += NTOK * 2 * D_INNER;
    float* xs       = ws;                     ws += NTOK * D_INNER;
    float* dbc      = ws;                     ws += NTOK * DBC_LD;
    float* dtpre    = ws;                     ws += NTOK * D_INNER;
    float* ybuf     = ws;                     ws += NTOK * D_INNER;
    float* fc       = ws;                     ws += NTOK * 2 * D_MLP;
    float* act      = ws;                     ws += NTOK * D_MLP;
    unsigned short* wbuf = (unsigned short*)ws;   // >= 4096*768 bf16 (6.3 MB)

    float* logits = (float*)d_out;
    float* loss   = (float*)d_out + (size_t)out_size - 1;

    embed_k<<<NTOK, 256, 0, stream>>>(idx, emb, hidden);

    for (int L = 0; L < NLAYER; ++L) {
        const float* const* P = (const float* const*)(d_in + 4 + 13 * L);
        const float* norm1_w = P[0];
        const float* in_proj = P[1];
        const float* conv_w  = P[2];
        const float* conv_b  = P[3];
        const float* x_proj  = P[4];
        const float* dt_w    = P[5];
        const float* dt_b    = P[6];
        const float* A_log   = P[7];
        const float* Dv      = P[8];
        const float* out_pr  = P[9];
        const float* norm2_w = P[10];
        const float* fc1     = P[11];
        const float* fc2     = P[12];

        add_rmsnorm_k<<<NTOK, 256, 0, stream>>>(hidden, residual, norm1_w, xn, L == 0);

        // xz = xn @ in_proj : N=3072, K=768
        cvtT(in_proj, wbuf, D_MODEL, 2 * D_INNER, D_MODEL, 2 * D_INNER, stream);
        gemm_wmma_t<false><<<dim3(2 * D_INNER / BN, NTOK / BM), 256, 0, stream>>>(
            xn, wbuf, xz, 2 * D_INNER, D_MODEL, D_MODEL, 2 * D_INNER);

        conv_silu_k<<<(NTOK * D_INNER + 255) / 256, 256, 0, stream>>>(xz, conv_w, conv_b, xs);

        // dbc = xs @ x_proj : N=80 padded to 128, K=1536
        cvtT(x_proj, wbuf, D_INNER, DT_RANK + 2 * D_STATE, D_INNER, DBC_LD, stream);
        gemm_wmma_t<false><<<dim3(DBC_LD / BN, NTOK / BM), 256, 0, stream>>>(
            xs, wbuf, dbc, DBC_LD, D_INNER, D_INNER, DBC_LD);

        // dtpre = dbc[:, :48] @ dt_w : K=48 padded to 64 (zero weight rows kill cols 48..63)
        cvtT(dt_w, wbuf, DT_RANK, D_INNER, 64, D_INNER, stream);
        gemm_wmma_t<false><<<dim3(D_INNER / BN, NTOK / BM), 256, 0, stream>>>(
            dbc, wbuf, dtpre, D_INNER, 64, DBC_LD, D_INNER);

        scan_k<<<dim3(D_INNER / 256, 2), 256, 0, stream>>>(
            dtpre, dt_b, A_log, xs, xz, dbc, Dv, ybuf);

        // hidden = y @ out_proj : N=768, K=1536
        cvtT(out_pr, wbuf, D_INNER, D_MODEL, D_INNER, D_MODEL, stream);
        gemm_wmma_t<false><<<dim3(D_MODEL / BN, NTOK / BM), 256, 0, stream>>>(
            ybuf, wbuf, hidden, D_MODEL, D_INNER, D_INNER, D_MODEL);

        add_rmsnorm_k<<<NTOK, 256, 0, stream>>>(hidden, residual, norm2_w, xn, 0);

        // fc = xn @ fc1 : N=4096, K=768
        cvtT(fc1, wbuf, D_MODEL, 2 * D_MLP, D_MODEL, 2 * D_MLP, stream);
        gemm_wmma_t<false><<<dim3(2 * D_MLP / BN, NTOK / BM), 256, 0, stream>>>(
            xn, wbuf, fc, 2 * D_MLP, D_MODEL, D_MODEL, 2 * D_MLP);

        glu_k<<<(NTOK * D_MLP + 255) / 256, 256, 0, stream>>>(fc, act);

        // hidden = act @ fc2 : N=768, K=2048
        cvtT(fc2, wbuf, D_MLP, D_MODEL, D_MLP, D_MODEL, stream);
        gemm_wmma_t<false><<<dim3(D_MODEL / BN, NTOK / BM), 256, 0, stream>>>(
            act, wbuf, hidden, D_MODEL, D_MLP, D_MLP, D_MODEL);
    }

    add_rmsnorm_k<<<NTOK, 256, 0, stream>>>(hidden, residual, norm_f, xn, 0);

    // logits = xn @ embedding^T : embedding is already [N=50304, K=768], f32 in-flight convert
    gemm_wmma_t<true><<<dim3(VOCAB / BN, NTOK / BM), 256, 0, stream>>>(
        xn, emb, logits, VOCAB, D_MODEL, D_MODEL, VOCAB);

    loss_zero_k<<<1, 32, 0, stream>>>(loss);
    loss_k<<<NTOK, 256, 0, stream>>>(logits, targets, loss);
}
